// VectorQuantizerEMA_78108275245195
// MI455X (gfx1250) — matrix operand order repaired and checked by reading, not compiled, not run on hardware
//
#include <hip/hip_runtime.h>
#include <hip/hip_bf16.h>

typedef __attribute__((ext_vector_type(16))) _Float16 v16h;
typedef __attribute__((ext_vector_type(8)))  _Float16 v8h;
typedef __attribute__((ext_vector_type(8)))  float    v8f;

#define M_TOT   16384          // 16*1024 flattened input rows
#define N_EMB   8192           // codebook entries
#define CDIM    256            // embedding dim
#define TOPK    3
#define NTILES  (N_EMB / 16)   // 512 column tiles
#define WAVES   8              // 256 threads / wave32

// ---------------------------------------------------------------- utilities
// Branchless sorted insert into ascending top-3 (val,idx): pure cmp/cndmask,
// no exec-mask divergence -> co-executes with the WMMA (XDL/TRANS) pipe.
__device__ __forceinline__ void insert3(float s, int id, float b[3], int bi[3]) {
    const bool c0 = s < b[0];
    const bool c1 = s < b[1];
    const bool c2 = s < b[2];
    float nb0 = c0 ? s     : b[0];
    int   ni0 = c0 ? id    : bi[0];
    float nb1 = c0 ? b[0]  : (c1 ? s  : b[1]);
    int   ni1 = c0 ? bi[0] : (c1 ? id : bi[1]);
    float nb2 = c1 ? b[1]  : (c2 ? s  : b[2]);
    int   ni2 = c1 ? bi[1] : (c2 ? id : bi[2]);
    b[0] = nb0; b[1] = nb1; b[2] = nb2;
    bi[0] = ni0; bi[1] = ni1; bi[2] = ni2;
}

// ---------------------------------------------------------------- kernels
__global__ void vq_zero(float* __restrict__ counts, float* __restrict__ sumdiff) {
    int i = blockIdx.x * 256 + threadIdx.x;
    if (i < N_EMB) counts[i] = 0.0f;
    if (i == 0)    *sumdiff = 0.0f;
}

__global__ void vq_convert(const float* __restrict__ x, const float* __restrict__ w,
                           _Float16* __restrict__ xh, _Float16* __restrict__ wh) {
    size_t i      = (size_t)blockIdx.x * blockDim.x + threadIdx.x;
    size_t stride = (size_t)gridDim.x * blockDim.x;
    const size_t MX = (size_t)M_TOT * CDIM;
    const size_t NW = (size_t)N_EMB * CDIM;
    for (size_t k = i; k < MX; k += stride) xh[k] = (_Float16)x[k];
    for (size_t k = i; k < NW; k += stride) wh[k] = (_Float16)w[k];
}

__global__ __launch_bounds__(256) void vq_wnorm(const float* __restrict__ w,
                                                float* __restrict__ wnorm) {
    __shared__ float red[256];
    const int row = blockIdx.x, t = threadIdx.x;
    float v = w[(size_t)row * CDIM + t];
    red[t] = v * v;
    __syncthreads();
    for (int s = 128; s > 0; s >>= 1) {
        if (t < s) red[t] += red[t + s];
        __syncthreads();
    }
    if (t == 0) wnorm[row] = red[0];
}

// Fused  score = ||w||^2 - 2 x.w  GEMM (WMMA f16 -> f32) + per-row top-3.
// Each wave processes TWO adjacent 16-col tiles per iteration: two independent
// WMMA accumulator chains (ILP on the matrix pipe) sharing the A fragments.
__global__ __launch_bounds__(256) void vq_topk(const _Float16* __restrict__ xh,
                                               const _Float16* __restrict__ wh,
                                               const float* __restrict__ wnorm,
                                               int* __restrict__ topidx) {
    const int m0   = blockIdx.x * 16;
    const int tid  = threadIdx.x;
    const int wave = tid >> 5;
    const int lane = tid & 31;
    const int lrow = lane & 15;
    const int hi   = lane >> 4;          // lane group 0 / 1

    // ---- preload A fragments: 16 rows x K=256, contiguous K per lane half ----
    // lane<16 holds K = kc*32 + 0..15; lane>=16 holds K = kc*32 + 16..31.
    v16h afrag[8];
    const _Float16* arow  = xh + (size_t)(m0 + lrow) * CDIM;
    const int       akoff = hi ? 16 : 0;
#pragma unroll
    for (int kc = 0; kc < 8; ++kc) {
        const _Float16* p  = arow + kc * 32 + akoff;
        v8h lo = *(const v8h*)(p);
        v8h hh = *(const v8h*)(p + 8);
        afrag[kc] = __builtin_shufflevector(lo, hh,
                       0, 1, 2, 3, 4, 5, 6, 7, 8, 9, 10, 11, 12, 13, 14, 15);
    }

    float best[8][3];
    int   besti[8][3];
#pragma unroll
    for (int v = 0; v < 8; ++v) {
        best[v][0] = best[v][1] = best[v][2] = 3.0e38f;
        besti[v][0] = besti[v][1] = besti[v][2] = 0;
    }

    // ---- sweep 512 column tiles, 2 tiles x 8 WMMAs (K=256) per iter ----
    // B layout: lane holds column n = lane%16, K contiguous (0..15 | 16..31).
    const int bkoff = hi ? 16 : 0;
    for (int base = 2 * wave; base < NTILES; base += 2 * WAVES) {
        const int n0 = base * 16;                    // 32 consecutive codebook rows
        const _Float16* brow0 = wh + (size_t)(n0 + lrow) * CDIM + bkoff;
        const _Float16* brow1 = brow0 + (size_t)16 * CDIM;
        v8f acc0 = {};
        v8f acc1 = {};
#pragma unroll
        for (int kc = 0; kc < 8; ++kc) {
            v16h b0 = *(const v16h*)(brow0 + kc * 32);
            v16h b1 = *(const v16h*)(brow1 + kc * 32);
            acc0 = __builtin_amdgcn_wmma_f32_16x16x32_f16(
                       false, afrag[kc], false, b0, (short)0, acc0, false, false);
            acc1 = __builtin_amdgcn_wmma_f32_16x16x32_f16(
                       false, afrag[kc], false, b1, (short)0, acc1, false, false);
        }
        const int   nc0 = n0 + lrow;
        const int   nc1 = nc0 + 16;
        const float wn0 = wnorm[nc0];
        const float wn1 = wnorm[nc1];
#pragma unroll
        for (int v = 0; v < 8; ++v) {
            insert3(fmaf(-2.0f, acc0[v], wn0), nc0, best[v], besti[v]);
            insert3(fmaf(-2.0f, acc1[v], wn1), nc1, best[v], besti[v]);
        }
    }

    // ---- merge: 8 waves x 16 lanes x 3 = 384 disjoint candidates per row ----
    __shared__ float sval[16][WAVES * 48];
    __shared__ int   sidx[16][WAVES * 48];
#pragma unroll
    for (int v = 0; v < 8; ++v) {
        const int row  = v + hi * 8;             // C-layout: VGPR v -> M = v (+8)
        const int base = wave * 48 + lrow * 3;
#pragma unroll
        for (int j = 0; j < 3; ++j) {
            sval[row][base + j] = best[v][j];
            sidx[row][base + j] = besti[v][j];
        }
    }
    __syncthreads();

    if (tid < 16) {
        float b[3]  = {3.0e38f, 3.0e38f, 3.0e38f};
        int   bi[3] = {0, 0, 0};
        for (int c = 0; c < WAVES * 48; ++c)
            insert3(sval[tid][c], sidx[tid][c], b, bi);
        int* out = topidx + (size_t)(m0 + tid) * TOPK;
        out[0] = bi[0]; out[1] = bi[1]; out[2] = bi[2];
    }
}

// Gather codebook rows -> quantized_st, accumulate diff^2, histogram counts.
__global__ __launch_bounds__(256) void vq_gather(const float* __restrict__ x,
                                                 const float* __restrict__ w,
                                                 const int* __restrict__ topidx,
                                                 float* __restrict__ quant_out,
                                                 float* __restrict__ idx_out,
                                                 float* __restrict__ counts,
                                                 float* __restrict__ sumdiff) {
    const int mk  = blockIdx.x;          // 0 .. M_TOT*TOPK-1
    const int m   = mk / TOPK;
    const int t   = threadIdx.x;
    const int idx = topidx[mk];

    float wv = w[(size_t)idx * CDIM + t];
    float xv = x[(size_t)m * CDIM + t];
    quant_out[(size_t)mk * CDIM + t] = wv;   // straight-through == quantized

    float d = wv - xv;
    d *= d;
    __shared__ float red[256];
    red[t] = d;
    __syncthreads();
    for (int s = 128; s > 0; s >>= 1) {
        if (t < s) red[t] += red[t + s];
        __syncthreads();
    }
    if (t == 0) {
        atomicAdd(sumdiff, red[0]);
        atomicAdd(&counts[idx], 1.0f);
        idx_out[mk] = (float)idx;
    }
}

__global__ __launch_bounds__(256) void vq_finalize(const float* __restrict__ counts,
                                                   const float* __restrict__ sumdiff,
                                                   float* __restrict__ loss_out,
                                                   float* __restrict__ perp_out) {
    __shared__ float red[256];
    const int t = threadIdx.x;
    float ent = 0.0f;
    for (int i = t; i < N_EMB; i += 256) {
        float p = counts[i] * (1.0f / (float)M_TOT);
        ent += p * __logf(p + 1e-10f);
    }
    red[t] = ent;
    __syncthreads();
    for (int s = 128; s > 0; s >>= 1) {
        if (t < s) red[t] += red[t + s];
        __syncthreads();
    }
    if (t == 0) {
        *perp_out = __expf(-red[0]);
        float mean = *sumdiff * (1.0f / ((float)M_TOT * TOPK * CDIM));
        *loss_out  = mean * (1.0f + 0.25f);   // q_latent + commitment*e_latent
    }
}

// ---------------------------------------------------------------- launch
extern "C" void kernel_launch(void* const* d_in, const int* in_sizes, int n_in,
                              void* d_out, int out_size, void* d_ws, size_t ws_size,
                              hipStream_t stream) {
    const float* x = (const float*)d_in[0];   // (16,1024,256) f32
    const float* w = (const float*)d_in[1];   // (8192,256)    f32

    // d_out: [loss | quantized_st (16*1024*3*256) | perplexity | indices (16*1024*3)]
    float* out       = (float*)d_out;
    float* loss_out  = out;
    float* quant_out = out + 1;
    float* perp_out  = out + 1 + (size_t)M_TOT * TOPK * CDIM;
    float* idx_out   = perp_out + 1;

    // workspace layout
    char*     ws      = (char*)d_ws;
    _Float16* xh      = (_Float16*)(ws);                 //  8 MB f16 inputs
    _Float16* wh      = (_Float16*)(ws + 8388608);       //  4 MB f16 codebook
    float*    wnorm   = (float*)   (ws + 12582912);      // 32 KB ||w||^2
    int*      topidx  = (int*)     (ws + 12615680);      // 192 KB top-3 indices
    float*    counts  = (float*)   (ws + 12812288);      // 32 KB histogram
    float*    sumdiff = (float*)   (ws + 12845056);      //  4 B  diff^2 accum

    vq_zero    <<<(N_EMB + 255) / 256, 256, 0, stream>>>(counts, sumdiff);
    vq_convert <<<2048,            256, 0, stream>>>(x, w, xh, wh);
    vq_wnorm   <<<N_EMB,           256, 0, stream>>>(w, wnorm);
    vq_topk    <<<M_TOT / 16,      256, 0, stream>>>(xh, wh, wnorm, topidx);
    vq_gather  <<<M_TOT * TOPK,    256, 0, stream>>>(x, w, topidx, quant_out,
                                                     idx_out, counts, sumdiff);
    vq_finalize<<<1,               256, 0, stream>>>(counts, sumdiff, loss_out, perp_out);
}
